// MultiHeadSelfAttention_61607010894462
// MI455X (gfx1250) — matrix operand order repaired and checked by reading, not compiled, not run on hardware
//
#include <hip/hip_runtime.h>
#include <hip/hip_bf16.h>

typedef __attribute__((ext_vector_type(16))) __bf16 v16bf;
typedef __attribute__((ext_vector_type(8)))  __bf16 v8bf;
typedef __attribute__((ext_vector_type(4)))  __bf16 v4bf;
typedef __attribute__((ext_vector_type(8)))  float  v8f;

union FragA {
    v16bf v;
    v8bf  h[2];
};

// ---------------------------------------------------------------------------
// gfx1250 async global->LDS copy (ASYNCcnt path).
// Per CDNA5 ISA: each lane copies bytes from its global address to its LDS address.
// LDS offset = low 32 bits of the generic shared-aperture address.
// ---------------------------------------------------------------------------
__device__ __forceinline__ unsigned lds_off(const void* p) {
    return (unsigned)(unsigned long long)p;
}
__device__ __forceinline__ void async_copy_b128(unsigned lds, const void* g) {
    asm volatile("global_load_async_to_lds_b128 %0, %1, off"
                 :: "v"(lds), "v"(g) : "memory");
}
__device__ __forceinline__ void wait_async0() {
    asm volatile("s_wait_asynccnt 0x0" ::: "memory");
}

// ---------------------------------------------------------------------------
// fp32 -> bf16 elementwise convert (vector of 4 per thread)
// ---------------------------------------------------------------------------
__global__ __launch_bounds__(256) void mhsa_cvt_bf16_kernel(
    const float* __restrict__ in, __bf16* __restrict__ out) {
    int i = (blockIdx.x * blockDim.x + threadIdx.x) * 4;
    float4 f = *(const float4*)(in + i);
    v4bf b;
    b[0] = (__bf16)f.x; b[1] = (__bf16)f.y; b[2] = (__bf16)f.z; b[3] = (__bf16)f.w;
    *(v4bf*)(out + i) = b;
}

// ---------------------------------------------------------------------------
// W[o][d] fp32 -> Wt[d][o] bf16  (so GEMM B-operand is row-major K x N)
// ---------------------------------------------------------------------------
__global__ __launch_bounds__(256) void mhsa_wtrans_kernel(
    const float* __restrict__ W, __bf16* __restrict__ Wt, int dim) {
    int d = blockIdx.x * 64 + threadIdx.x;   // blockDim = (64,4)
    int o = blockIdx.y * 4 + threadIdx.y;
    Wt[(size_t)d * dim + o] = (__bf16)W[(size_t)o * dim + d];
}

// ---------------------------------------------------------------------------
// bf16 WMMA GEMM: C[M,N] = A[M,K] * B[K,N], A/B row-major bf16, f32 accumulate.
// Block tile 128x128, 8 waves (wave32), wave tile 32x64 (2x4 WMMA), K-step 32.
// A tile staged with async global->LDS; B tile staged transposed via VALU.
// ---------------------------------------------------------------------------
template <bool F32OUT>
__global__ __launch_bounds__(256) void mhsa_gemm_bf16_kernel(
    const __bf16* __restrict__ A, const __bf16* __restrict__ B,
    void* __restrict__ Cout, int M, int N, int K) {
    __shared__ __align__(32) __bf16 As[128][40];    // [m][k], padded
    __shared__ __align__(32) __bf16 Bst[128][48];   // [n][k], padded (B staged transposed)

    const int tid  = threadIdx.x;
    const int wave = tid >> 5;
    const int lane = tid & 31;
    const int hi   = (lane >> 4) & 1;
    const int l16  = lane & 15;
    const int wm   = wave >> 1;   // 0..3 -> M strip (32 rows)
    const int wn   = wave & 1;    // 0..1 -> N strip (64 cols)
    const int m0   = blockIdx.y * 128;
    const int n0   = blockIdx.x * 128;

    v8f acc[2][4] = {};

    for (int k0 = 0; k0 < K; k0 += 32) {
        __syncthreads();
        // stage A tile 128x32 via async copies: 2 threads per row, 16 elems each
        {
            int row = tid >> 1;
            int c0  = (tid & 1) * 16;
            const __bf16* g = A + (size_t)(m0 + row) * K + k0 + c0;
            unsigned l = lds_off(&As[row][c0]);
            async_copy_b128(l,      g);
            async_copy_b128(l + 16, g + 8);
        }
        // stage B tile 32x128 transposed into Bst[n][k]
        #pragma unroll
        for (int it = 0; it < 4; ++it) {
            int u  = it * 256 + tid;      // 0..1023 units of 4 elems
            int kk = u >> 5;              // 0..31
            int nn = (u & 31) * 4;        // 0..127
            v4bf t = *(const v4bf*)(B + (size_t)(k0 + kk) * N + n0 + nn);
            #pragma unroll
            for (int j = 0; j < 4; ++j) Bst[nn + j][kk] = t[j];
        }
        wait_async0();
        __syncthreads();

        // A fragments: 16x32, contiguous k order, same as B (row = l16, k split by half-wave)
        FragA a[2];
        #pragma unroll
        for (int sm = 0; sm < 2; ++sm) {
            int row = wm * 32 + sm * 16 + l16;
            a[sm].h[0] = *(const v8bf*)(&As[row][hi * 16]);       // k = 0..7   (+16 if hi)
            a[sm].h[1] = *(const v8bf*)(&As[row][hi * 16 + 8]);   // k = 8..15  (+16 if hi)
        }
        // B fragments: 32x16, n = l16, k = j + 16*hi  -> contiguous in Bst[n][*]
        FragA b[4];
        #pragma unroll
        for (int sn = 0; sn < 4; ++sn) {
            int col = wn * 64 + sn * 16 + l16;
            const v8bf* p = (const v8bf*)(&Bst[col][hi * 16]);
            b[sn].h[0] = p[0];
            b[sn].h[1] = p[1];
        }
        #pragma unroll
        for (int sm = 0; sm < 2; ++sm)
            #pragma unroll
            for (int sn = 0; sn < 4; ++sn)
                acc[sm][sn] = __builtin_amdgcn_wmma_f32_16x16x32_bf16(
                    false, a[sm].v, false, b[sn].v, (short)0, acc[sm][sn], false, false);
    }

    // store: C/D layout row = r + 8*hi, col = l16
    #pragma unroll
    for (int sm = 0; sm < 2; ++sm) {
        int mbase = m0 + wm * 32 + sm * 16 + hi * 8;
        #pragma unroll
        for (int sn = 0; sn < 4; ++sn) {
            int col = n0 + wn * 64 + sn * 16 + l16;
            #pragma unroll
            for (int r = 0; r < 8; ++r) {
                size_t off = (size_t)(mbase + r) * N + col;
                if (F32OUT) ((float*)Cout)[off] = acc[sm][sn][r];
                else        ((__bf16*)Cout)[off] = (__bf16)acc[sm][sn][r];
            }
        }
    }
}

// ---------------------------------------------------------------------------
// Flash attention (causal): one block = 64 queries x 1 head, 128 threads
// (4 waves, each owning a 16-query strip). Online softmax, bf16 WMMA.
// K block staged with async global->LDS; V staged transposed via VALU.
// ---------------------------------------------------------------------------
__global__ __launch_bounds__(128) void mhsa_flash_attn_kernel(
    const __bf16* __restrict__ Q, const __bf16* __restrict__ Kg,
    const __bf16* __restrict__ Vg, __bf16* __restrict__ ctx,
    int S, int Dm) {
    const int h    = blockIdx.y;
    const int qb   = blockIdx.x;           // 64-query block
    const int tid  = threadIdx.x;
    const int wave = tid >> 5;             // 0..3
    const int lane = tid & 31;
    const int hi   = (lane >> 4) & 1;
    const int l16  = lane & 15;
    const float scale = 0.125f;            // 1/sqrt(64)

    __shared__ __align__(32) __bf16 Ks[64][80];       // [key][d]   (B for Q*K^T)
    __shared__ __align__(32) __bf16 Vt[64][80];       // [d][key]   (B for P*V)
    __shared__ __align__(32) float  Pls[4][16][68];   // per-wave P relayout scratch

    // Q fragments: A 16x64 -> two K=32 chunks
    const int qrow = qb * 64 + wave * 16 + l16;
    FragA qf[2];
    #pragma unroll
    for (int c = 0; c < 2; ++c) {
        const __bf16* qp = Q + (size_t)qrow * Dm + h * 64 + c * 32 + hi * 16;
        qf[c].h[0] = *(const v8bf*)qp;
        qf[c].h[1] = *(const v8bf*)(qp + 8);
    }

    v8f oacc[4] = {};
    float mrow[8], lrow[8];
    #pragma unroll
    for (int r = 0; r < 8; ++r) { mrow[r] = -1e30f; lrow[r] = 0.0f; }

    for (int kb = 0; kb <= qb; ++kb) {
        __syncthreads();
        // stage K block [key][d] via async: 2 threads per key row, 32 d each
        {
            int row = tid >> 1;
            int c0  = (tid & 1) * 32;
            const __bf16* g = Kg + (size_t)(kb * 64 + row) * Dm + h * 64 + c0;
            unsigned l = lds_off(&Ks[row][c0]);
            async_copy_b128(l,      g);
            async_copy_b128(l + 16, g + 8);
            async_copy_b128(l + 32, g + 16);
            async_copy_b128(l + 48, g + 24);
        }
        // stage V block transposed [d][key]
        #pragma unroll
        for (int it = 0; it < 8; ++it) {
            int u   = it * 128 + tid;      // 0..1023 units of 4 elems
            int key = u >> 4;              // 0..63
            int d0  = (u & 15) * 4;        // 0..63
            v4bf t = *(const v4bf*)(Vg + (size_t)(kb * 64 + key) * Dm + h * 64 + d0);
            #pragma unroll
            for (int j = 0; j < 4; ++j) Vt[d0 + j][key] = t[j];
        }
        wait_async0();
        __syncthreads();

        // scores: S = Q * K^T over 4 key tiles of 16
        v8f sc[4];
        #pragma unroll
        for (int t = 0; t < 4; ++t) {
            FragA kf0, kf1;
            int col = t * 16 + l16;                                // key index in block
            const v8bf* p0 = (const v8bf*)(&Ks[col][hi * 16]);     // d chunk 0
            kf0.h[0] = p0[0]; kf0.h[1] = p0[1];
            const v8bf* p1 = (const v8bf*)(&Ks[col][32 + hi * 16]);// d chunk 1
            kf1.h[0] = p1[0]; kf1.h[1] = p1[1];
            v8f c = {};
            c = __builtin_amdgcn_wmma_f32_16x16x32_bf16(false, qf[0].v, false, kf0.v, (short)0, c, false, false);
            c = __builtin_amdgcn_wmma_f32_16x16x32_bf16(false, qf[1].v, false, kf1.v, (short)0, c, false, false);
            sc[t] = c;
        }

        // scale + causal mask
        #pragma unroll
        for (int t = 0; t < 4; ++t) {
            #pragma unroll
            for (int r = 0; r < 8; ++r) {
                int qg = qb * 64 + wave * 16 + r + hi * 8;
                int kg = kb * 64 + t * 16 + l16;
                float s = sc[t][r] * scale;
                sc[t][r] = (kg <= qg) ? s : -1e30f;
            }
        }

        // online softmax: row max (rows live in 16-lane halves)
        float mnew[8], alpha[8];
        #pragma unroll
        for (int r = 0; r < 8; ++r) {
            float m = fmaxf(fmaxf(sc[0][r], sc[1][r]), fmaxf(sc[2][r], sc[3][r]));
            #pragma unroll
            for (int off = 8; off >= 1; off >>= 1)
                m = fmaxf(m, __shfl_xor(m, off, 32));
            mnew[r]  = fmaxf(mrow[r], m);
            alpha[r] = __expf(mrow[r] - mnew[r]);
            mrow[r]  = mnew[r];
        }
        // exp + row sum
        #pragma unroll
        for (int t = 0; t < 4; ++t)
            #pragma unroll
            for (int r = 0; r < 8; ++r)
                sc[t][r] = __expf(sc[t][r] - mnew[r]);
        #pragma unroll
        for (int r = 0; r < 8; ++r) {
            float su = sc[0][r] + sc[1][r] + sc[2][r] + sc[3][r];
            #pragma unroll
            for (int off = 8; off >= 1; off >>= 1)
                su += __shfl_xor(su, off, 32);
            lrow[r] = lrow[r] * alpha[r] + su;
        }
        // rescale running output
        #pragma unroll
        for (int dt = 0; dt < 4; ++dt)
            #pragma unroll
            for (int r = 0; r < 8; ++r)
                oacc[dt][r] *= alpha[r];

        // relayout P: D-format f32 -> A-format bf16 via per-wave LDS scratch
        // (same-wave DS ops are in order; no barrier needed)
        #pragma unroll
        for (int t = 0; t < 4; ++t)
            #pragma unroll
            for (int r = 0; r < 8; ++r)
                Pls[wave][r + hi * 8][t * 16 + l16] = sc[t][r];
        FragA pf[2];
        #pragma unroll
        for (int c = 0; c < 2; ++c) {
            #pragma unroll
            for (int j = 0; j < 16; ++j) {
                int k = j + hi * 16 + c * 32;
                pf[c].v[j] = (__bf16)Pls[wave][l16][k];
            }
        }

        // O += P * V over 4 d-tiles
        #pragma unroll
        for (int dt = 0; dt < 4; ++dt) {
            FragA vf0, vf1;
            int dcol = dt * 16 + l16;
            const v8bf* p0 = (const v8bf*)(&Vt[dcol][hi * 16]);
            vf0.h[0] = p0[0]; vf0.h[1] = p0[1];
            const v8bf* p1 = (const v8bf*)(&Vt[dcol][32 + hi * 16]);
            vf1.h[0] = p1[0]; vf1.h[1] = p1[1];
            oacc[dt] = __builtin_amdgcn_wmma_f32_16x16x32_bf16(false, pf[0].v, false, vf0.v, (short)0, oacc[dt], false, false);
            oacc[dt] = __builtin_amdgcn_wmma_f32_16x16x32_bf16(false, pf[1].v, false, vf1.v, (short)0, oacc[dt], false, false);
        }
    }

    // normalize + store context (bf16, feeds final O-projection GEMM)
    #pragma unroll
    for (int dt = 0; dt < 4; ++dt) {
        #pragma unroll
        for (int r = 0; r < 8; ++r) {
            int q = qb * 64 + wave * 16 + r + hi * 8;
            float o = oacc[dt][r] / lrow[r];
            ctx[(size_t)q * Dm + h * 64 + dt * 16 + l16] = (__bf16)o;
        }
    }
}

// ---------------------------------------------------------------------------
// launch
// ---------------------------------------------------------------------------
extern "C" void kernel_launch(void* const* d_in, const int* in_sizes, int n_in,
                              void* d_out, int out_size, void* d_ws, size_t ws_size,
                              hipStream_t stream) {
    (void)in_sizes; (void)n_in; (void)out_size; (void)ws_size;
    const int S = 4096, D = 1024, H = 16;
    const float* x = (const float*)d_in[0];
    const float* W[4] = { (const float*)d_in[1], (const float*)d_in[2],
                          (const float*)d_in[3], (const float*)d_in[4] }; // Q,K,V,O
    float* out = (float*)d_out;

    const size_t SD = (size_t)S * D;
    char* p = (char*)d_ws;
    __bf16* xb = (__bf16*)p;  p += SD * 2;
    __bf16* Wt[4];
    for (int i = 0; i < 4; ++i) { Wt[i] = (__bf16*)p; p += (size_t)D * D * 2; }
    __bf16* Qb  = (__bf16*)p; p += SD * 2;
    __bf16* Kb  = (__bf16*)p; p += SD * 2;
    __bf16* Vb  = (__bf16*)p; p += SD * 2;
    __bf16* Ctx = (__bf16*)p; p += SD * 2;

    // 1) convert inputs to bf16 (weights transposed so GEMM B is row-major KxN)
    mhsa_cvt_bf16_kernel<<<dim3((unsigned)(SD / 1024)), dim3(256), 0, stream>>>(x, xb);
    for (int i = 0; i < 4; ++i)
        mhsa_wtrans_kernel<<<dim3(D / 64, D / 4), dim3(64, 4), 0, stream>>>(W[i], Wt[i], D);

    // 2) Q/K/V projections
    dim3 ggrid(D / 128, S / 128);
    mhsa_gemm_bf16_kernel<false><<<ggrid, dim3(256), 0, stream>>>(xb, Wt[0], Qb, S, D, D);
    mhsa_gemm_bf16_kernel<false><<<ggrid, dim3(256), 0, stream>>>(xb, Wt[1], Kb, S, D, D);
    mhsa_gemm_bf16_kernel<false><<<ggrid, dim3(256), 0, stream>>>(xb, Wt[2], Vb, S, D, D);

    // 3) causal flash attention per (query-block, head)
    mhsa_flash_attn_kernel<<<dim3(S / 64, H), dim3(128), 0, stream>>>(Qb, Kb, Vb, Ctx, S, D);

    // 4) output projection (f32 out)
    mhsa_gemm_bf16_kernel<true><<<ggrid, dim3(256), 0, stream>>>(Ctx, Wt[3], out, S, D, D);
}
